// LSTMEncoder_30872224924229
// MI455X (gfx1250) — matrix operand order repaired and checked by reading, not compiled, not run on hardware
//
#include <hip/hip_runtime.h>
#include <hip/hip_bf16.h>

// Problem constants (from reference): B=64, T=1024, D=512, H=1024, LAT=256
#define BB   64
#define TT   1024
#define DD   512
#define HH   1024
#define G4   4096      // 4*H gate rows
#define KK   1536      // D + H combined K
#define LATN 256
#define WPAD 1032      // LDS row stride in shorts (1024 + 8): stride%64banks==4

typedef __attribute__((ext_vector_type(8)))  float  v8f;
typedef __attribute__((ext_vector_type(16))) __bf16 v16bf;

union BFrag {
    v16bf v;
    uint4 q[2];
};

__device__ __forceinline__ unsigned short f2bf(float f) {
    union { float f; unsigned u; } x; x.f = f;
    unsigned r = ((x.u >> 16) & 1u) + 0x7FFFu;   // round-to-nearest-even
    return (unsigned short)((x.u + r) >> 16);
}
__device__ __forceinline__ unsigned pack2(float a, float b) {
    return (unsigned)f2bf(a) | ((unsigned)f2bf(b) << 16);
}
__device__ __forceinline__ float bf2f(unsigned short u) {
    union { unsigned u; float f; } x; x.u = ((unsigned)u) << 16;
    return x.f;
}
__device__ __forceinline__ float sigmoidf_fast(float x) {
    return 1.0f / (1.0f + __expf(-x));
}
__device__ __forceinline__ float tanhf_fast(float x) {
    x = fminf(fmaxf(x, -15.0f), 15.0f);
    float e = __expf(2.0f * x);
    return (e - 1.0f) / (e + 1.0f);
}

// Load one K-chunk fragment set (A + 4 gate B-frags) from global memory.
__device__ __forceinline__ void load_chunk(BFrag& a, BFrag (&bf)[4],
                                           const unsigned short* __restrict__ arow,
                                           const unsigned short* const (&wr)[4],
                                           int koff)
{
    a.q[0] = *(const uint4*)(arow + koff);
    a.q[1] = *(const uint4*)(arow + koff + 16);
    #pragma unroll
    for (int g = 0; g < 4; ++g) {
        const uint4* wp = (const uint4*)(wr[g] + koff);
        bf[g].q[0] = wp[0];
        bf[g].q[1] = wp[2];
    }
}

// ---------------------------------------------------------------------------
// Prep: permuted, gate-interleaved, bf16 combined weights.
// Permuted row n': hb=n'>>6, pos=n'&63, gate=pos>>4, jloc=pos&15,
//                  orig = gate*H + hb*16 + jloc
// Wc[n'] = [ w_ih[orig][0:512] | w_hh[orig][0:1024] ]  (bf16)
// ---------------------------------------------------------------------------
__global__ void prep_weights_kernel(const float* __restrict__ w_ih,
                                    const float* __restrict__ w_hh,
                                    const float* __restrict__ bias,
                                    unsigned short* __restrict__ Wc,
                                    float* __restrict__ bperm)
{
    const int np   = blockIdx.x;            // 0..4095
    const int hb   = np >> 6;
    const int pos  = np & 63;
    const int gate = pos >> 4;
    const int jloc = pos & 15;
    const int orig = gate * HH + hb * 16 + jloc;

    const float* sih = w_ih + (size_t)orig * DD;
    const float* shh = w_hh + (size_t)orig * HH;
    unsigned short* dst = Wc + (size_t)np * KK;

    for (int k = threadIdx.x; k < DD; k += blockDim.x) dst[k]      = f2bf(sih[k]);
    for (int k = threadIdx.x; k < HH; k += blockDim.x) dst[DD + k] = f2bf(shh[k]);
    if (threadIdx.x == 0) bperm[np] = bias[orig];
}

__global__ void init_state_kernel(unsigned short* __restrict__ h0,
                                  float* __restrict__ c)
{
    int i = blockIdx.x * blockDim.x + threadIdx.x;
    if (i < BB * HH) { h0[i] = 0; c[i] = 0.0f; }
}

// x fp32 -> bf16, one shot (bandwidth bound, ~200 MB total traffic)
__global__ void cvt_x_kernel(const float* __restrict__ x,
                             unsigned short* __restrict__ xbf)
{
    int i = blockIdx.x * blockDim.x + threadIdx.x;      // 8 elements per thread
    if (i >= (BB * TT * DD) / 8) return;
    const float4* p = (const float4*)x + (size_t)i * 2;
    float4 a = p[0], b = p[1];
    uint4 o;
    o.x = pack2(a.x, a.y); o.y = pack2(a.z, a.w);
    o.z = pack2(b.x, b.y); o.w = pack2(b.z, b.w);
    ((uint4*)xbf)[i] = o;
}

// ---------------------------------------------------------------------------
// Hoisted input projection: xproj[t] = x_t @ w_ih^T + b for all t (parallel
// big GEMM, M=B*T=65536, N=4096, K=512), stored in the WMMA D-tile layout
// [t][nb][w][g][lane][r] so the recurrent kernel can v8f-load accumulators.
// Grid: (TT/4, 64); block 512 threads = 16 waves (4 t-values x 4 m-strips).
// High occupancy hides latency here.
// ---------------------------------------------------------------------------
__global__ __launch_bounds__(512, 1)
void xproj_kernel(const unsigned short* __restrict__ xbf,  // [B][T][D] bf16
                  const unsigned short* __restrict__ Wc,   // [4096][1536] bf16 permuted
                  const float* __restrict__ bperm,         // [4096]
                  float* __restrict__ xproj)               // [T][64][4][4][32][8] f32
{
    const int lane = threadIdx.x & 31;
    const int wv   = threadIdx.x >> 5;   // 0..15
    const int w    = wv & 3;             // batch strip
    const int tsub = wv >> 2;            // 0..3
    const int t    = blockIdx.x * 4 + tsub;
    const int nb   = blockIdx.y;
    const int l15  = lane & 15;
    const int lhi  = lane >> 4;

    const int mA    = w * 16 + l15;
    const int kbA   = lhi * 8;
    const int col0  = nb * 64;
    const int koffB = lhi * 8;

    const unsigned short* wr[4];
    v8f acc[4];
    #pragma unroll
    for (int g = 0; g < 4; ++g) {
        int n = col0 + g * 16 + l15;
        wr[g] = Wc + (size_t)n * KK + koffB;
        float bv = bperm[n];
        acc[g] = (v8f){bv, bv, bv, bv, bv, bv, bv, bv};
    }

    const unsigned short* xrow = xbf + ((size_t)mA * TT + (size_t)t) * DD + kbA;
    #pragma unroll
    for (int kc = 0; kc < 16; ++kc) {
        BFrag a, b[4];
        load_chunk(a, b, xrow, wr, kc * 32);
        #pragma unroll
        for (int g = 0; g < 4; ++g)
            acc[g] = __builtin_amdgcn_wmma_f32_16x16x32_bf16(
                false, a.v, false, b[g].v, (short)0, acc[g], false, false);
    }

    float* xp = xproj + ((((size_t)t * 64 + nb) * 16 + w * 4) * 32 + lane) * 8;
    #pragma unroll
    for (int g = 0; g < 4; ++g)
        *(v8f*)(xp + g * 256) = acc[g];
}

// ---------------------------------------------------------------------------
// One LSTM time step. Grid: 64 blocks x 256 threads (8 waves).
// wave wv: w = wv&3 -> batch strip m in [16w,16w+16); half = wv>>2 -> K slice.
// MODE 2 (preferred): accumulators seeded from precomputed xproj; the block's
//   64x1024 bf16 recurrent-weight slice is cooperatively staged into LDS
//   (independent, pipelined global loads), then the serial WMMA loop reads
//   B-frags via ds_load_b128 (short LDS latency instead of exposed L2 latency)
//   with A double-buffered from global. Rows padded to 1032 shorts for
//   conflict-free gate-tile access.
// MODE 0/1: fallbacks reading x inline / bf16-x from global.
// Half 1 reduces into half 0 via LDS; half 0 does the i/f/g/o update
// register-locally (gate-interleaved weight permutation).
// ---------------------------------------------------------------------------
template <int MODE>
__global__ __launch_bounds__(256)
__attribute__((amdgpu_waves_per_eu(2, 2)))
void lstm_step_kernel(const float* __restrict__ x,            // [B][T][D] fp32
                      const unsigned short* __restrict__ xbf, // [B][T][D] bf16
                      const float* __restrict__ xproj,        // tile layout (MODE 2)
                      const unsigned short* __restrict__ Wc,  // [4096][1536] bf16 permuted
                      const float* __restrict__ bperm,        // [4096]
                      const unsigned short* __restrict__ hin, // [B][H] bf16
                      unsigned short* __restrict__ hout,      // [B][H] bf16
                      float* __restrict__ c,                  // [B][H] fp32
                      int t)
{
    __shared__ __align__(32) float lred[4 * 1024];   // 16 KB reduction buffer

    const int lane = threadIdx.x & 31;
    const int wv   = threadIdx.x >> 5;   // 0..7
    const int w    = wv & 3;             // batch strip
    const int half = wv >> 2;            // K-split index 0..1
    const int l15  = lane & 15;
    const int lhi  = lane >> 4;

    const int m0    = w * 16;
    const int mA    = m0 + l15;          // A row (lanes 0-15 & 16-31 both M=0..15)
    const int kbA   = lhi * 8;           // A K sub-base (hi lanes -> K+8)
    const int col0  = blockIdx.x * 64;
    const int koffB = lhi * 8;           // B K sub-base (hi lanes -> K+8)

    v8f acc[4];
    #pragma unroll
    for (int g = 0; g < 4; ++g) {
        float bv = 0.0f;
        if (MODE != 2 && half == 0) bv = bperm[col0 + g * 16 + l15];
        acc[g] = (v8f){bv, bv, bv, bv, bv, bv, bv, bv};
    }

    if constexpr (MODE == 2) {
        // ---- cooperative stage: block's 64x1024 h-part weight slice -> LDS ----
        __shared__ __align__(16) unsigned short smemW[64 * WPAD];   // 132 KB
        for (int idx = threadIdx.x; idx < 64 * 128; idx += 256) {
            const int r = idx >> 7;        // row in slice
            const int q = idx & 127;       // uint4 within row
            *(uint4*)(&smemW[r * WPAD + q * 8]) =
                *(const uint4*)(Wc + (size_t)(col0 + r) * KK + DD + q * 8);
        }
        __syncthreads();

        // seed accumulators with hoisted x-projection (bias already folded in)
        if (half == 0) {
            const float* xp = xproj +
                ((((size_t)t * 64 + blockIdx.x) * 16 + w * 4) * 32 + lane) * 8;
            #pragma unroll
            for (int g = 0; g < 4; ++g)
                acc[g] = *(const v8f*)(xp + g * 256);
        }

        // ---- 32 h chunks split 2 ways; B from LDS, A double-buffered ----
        const int kc0 = half * 16;
        const unsigned short* hrow = hin + (size_t)mA * HH + kbA;
        BFrag a[2];
        a[0].q[0] = *(const uint4*)(hrow + kc0 * 32);
        a[0].q[1] = *(const uint4*)(hrow + kc0 * 32 + 16);
        #pragma unroll
        for (int i = 0; i < 16; ++i) {
            const int cur = i & 1, nxt = cur ^ 1;
            if (i < 15) {
                const int kn = (kc0 + i + 1) * 32;
                a[nxt].q[0] = *(const uint4*)(hrow + kn);
                a[nxt].q[1] = *(const uint4*)(hrow + kn + 16);
            }
            const int kh = (kc0 + i) * 32;
            #pragma unroll
            for (int g = 0; g < 4; ++g) {
                BFrag b;
                const uint4* wp = (const uint4*)(&smemW[(g * 16 + l15) * WPAD + koffB + kh]);
                b.q[0] = wp[0];
                b.q[1] = wp[2];
                acc[g] = __builtin_amdgcn_wmma_f32_16x16x32_bf16(
                    false, a[cur].v, false, b.v, (short)0, acc[g], false, false);
            }
        }
    } else {
        // ---- 48 chunks (16 x-chunks then 32 h-chunks) split 2 ways ----
        const unsigned short* wrow[4];
        #pragma unroll
        for (int g = 0; g < 4; ++g)
            wrow[g] = Wc + (size_t)(col0 + g * 16 + l15) * KK + koffB;

        const int kc0 = half * 24;
        const int kc1 = kc0 + 24;
        const int xc_end = (kc1 < 16) ? kc1 : 16;
        if (MODE == 1) {
            const unsigned short* xrow = xbf + ((size_t)mA * TT + (size_t)t) * DD + kbA;
            for (int kc = kc0; kc < xc_end; ++kc) {
                BFrag a, b[4];
                load_chunk(a, b, xrow, wrow, kc * 32);
                #pragma unroll
                for (int g = 0; g < 4; ++g)
                    acc[g] = __builtin_amdgcn_wmma_f32_16x16x32_bf16(
                        false, a.v, false, b[g].v, (short)0, acc[g], false, false);
            }
        } else {
            const float* xrow = x + ((size_t)mA * TT + (size_t)t) * DD + kbA;
            for (int kc = kc0; kc < xc_end; ++kc) {
                const int k0 = kc * 32;
                BFrag a;
                float4 f0 = *(const float4*)(xrow + k0);
                float4 f1 = *(const float4*)(xrow + k0 + 4);
                float4 f2 = *(const float4*)(xrow + k0 + 16);
                float4 f3 = *(const float4*)(xrow + k0 + 20);
                a.q[0] = make_uint4(pack2(f0.x, f0.y), pack2(f0.z, f0.w),
                                    pack2(f1.x, f1.y), pack2(f1.z, f1.w));
                a.q[1] = make_uint4(pack2(f2.x, f2.y), pack2(f2.z, f2.w),
                                    pack2(f3.x, f3.y), pack2(f3.z, f3.w));
                #pragma unroll
                for (int g = 0; g < 4; ++g) {
                    BFrag b;
                    const uint4* wp = (const uint4*)(wrow[g] + k0);
                    b.q[0] = wp[0]; b.q[1] = wp[2];
                    acc[g] = __builtin_amdgcn_wmma_f32_16x16x32_bf16(
                        false, a.v, false, b.v, (short)0, acc[g], false, false);
                }
            }
        }
        const unsigned short* hrow = hin + (size_t)mA * HH + kbA;
        const int hc_begin = (kc0 > 16) ? kc0 : 16;
        for (int kc = hc_begin; kc < kc1; ++kc) {
            const int kh = (kc - 16) * 32;
            const int kw = kc * 32;
            BFrag a, b[4];
            a.q[0] = *(const uint4*)(hrow + kh);
            a.q[1] = *(const uint4*)(hrow + kh + 16);
            #pragma unroll
            for (int g = 0; g < 4; ++g) {
                const uint4* wp = (const uint4*)(wrow[g] + kw);
                b[g].q[0] = wp[0]; b[g].q[1] = wp[2];
            }
            #pragma unroll
            for (int g = 0; g < 4; ++g)
                acc[g] = __builtin_amdgcn_wmma_f32_16x16x32_bf16(
                    false, a.v, false, b[g].v, (short)0, acc[g], false, false);
        }
    }

    // ---- cross-half reduction through LDS ----
    if (half != 0) {
        float* dst = &lred[(w * 32 + lane) * 32];
        #pragma unroll
        for (int g = 0; g < 4; ++g)
            *(v8f*)(dst + g * 8) = acc[g];
    }
    __syncthreads();
    if (half == 0) {
        {
            const float* src = &lred[(w * 32 + lane) * 32];
            #pragma unroll
            for (int g = 0; g < 4; ++g)
                acc[g] += *(const v8f*)(src + g * 8);
        }
        // pointwise LSTM update: i/f/g/o live in same lane & register slot
        const int j = blockIdx.x * 16 + l15;
        #pragma unroll
        for (int r = 0; r < 8; ++r) {
            int m = m0 + lhi * 8 + r;          // D tile: lanes 0-15 M=r, 16-31 M=8+r
            float iv = sigmoidf_fast(acc[0][r]);
            float fv = sigmoidf_fast(acc[1][r]);
            float gv = tanhf_fast(acc[2][r]);
            float ov = sigmoidf_fast(acc[3][r]);
            size_t idx = (size_t)m * HH + j;
            float cn = fv * c[idx] + iv * gv;
            c[idx]   = cn;
            hout[idx] = f2bf(ov * tanhf_fast(cn));
        }
    }
}

// ---------------------------------------------------------------------------
// Final linear: out = h_final @ w_lin^T + b_lin ; split into mu | logvar.
// ---------------------------------------------------------------------------
__global__ void final_linear_kernel(const unsigned short* __restrict__ h,
                                    const float* __restrict__ w_lin,
                                    const float* __restrict__ b_lin,
                                    float* __restrict__ out)
{
    int idx = blockIdx.x * blockDim.x + threadIdx.x;
    if (idx >= BB * 2 * LATN) return;
    int b = idx / (2 * LATN);
    int l = idx % (2 * LATN);
    const unsigned short* hr = h + (size_t)b * HH;
    const float* wr = w_lin + (size_t)l * HH;
    float s = b_lin[l];
    #pragma unroll 4
    for (int k = 0; k < HH; ++k) s = fmaf(bf2f(hr[k]), wr[k], s);
    if (l < LATN) out[b * LATN + l] = s;
    else          out[BB * LATN + b * LATN + (l - LATN)] = s;
}

// ---------------------------------------------------------------------------
extern "C" void kernel_launch(void* const* d_in, const int* in_sizes, int n_in,
                              void* d_out, int out_size, void* d_ws, size_t ws_size,
                              hipStream_t stream)
{
    (void)in_sizes; (void)n_in; (void)out_size;

    const float* x     = (const float*)d_in[0];
    const float* w_ih  = (const float*)d_in[1];
    const float* w_hh  = (const float*)d_in[2];
    const float* b     = (const float*)d_in[3];
    const float* w_lin = (const float*)d_in[4];
    const float* b_lin = (const float*)d_in[5];
    float* out = (float*)d_out;

    // Workspace layout
    char* ws = (char*)d_ws;
    size_t off = 0;
    unsigned short* Wc = (unsigned short*)(ws + off); off += (size_t)G4 * KK * 2;  // 12,582,912
    float* bperm       = (float*)(ws + off);          off += (size_t)G4 * 4;       //     16,384
    unsigned short* h0 = (unsigned short*)(ws + off); off += (size_t)BB * HH * 2;  //    131,072
    unsigned short* h1 = (unsigned short*)(ws + off); off += (size_t)BB * HH * 2;  //    131,072
    float* c           = (float*)(ws + off);          off += (size_t)BB * HH * 4;  //    262,144
    unsigned short* xbf = (unsigned short*)(ws + off);
    const size_t need_xbf   = off + (size_t)BB * TT * DD * 2;          // + 67,108,864
    float* xproj = (float*)(ws + need_xbf);
    const size_t need_xproj = need_xbf + (size_t)TT * G4 * BB * 4;     // + 1,073,741,824

    // deterministic mode selection from ws_size only
    const int mode = (ws_size >= need_xproj) ? 2 : (ws_size >= need_xbf) ? 1 : 0;

    prep_weights_kernel<<<G4, 256, 0, stream>>>(w_ih, w_hh, b, Wc, bperm);
    init_state_kernel<<<(BB * HH + 255) / 256, 256, 0, stream>>>(h0, c);
    if (mode >= 1) {
        cvt_x_kernel<<<(BB * TT * DD / 8 + 255) / 256, 256, 0, stream>>>(x, xbf);
    }
    if (mode == 2) {
        xproj_kernel<<<dim3(TT / 4, 64), 512, 0, stream>>>(xbf, Wc, bperm, xproj);
    }

    unsigned short* hbuf[2] = {h0, h1};
    if (mode == 2) {
        for (int t = 0; t < TT; ++t)
            lstm_step_kernel<2><<<64, 256, 0, stream>>>(
                x, xbf, xproj, Wc, bperm, hbuf[t & 1], hbuf[(t + 1) & 1], c, t);
    } else if (mode == 1) {
        for (int t = 0; t < TT; ++t)
            lstm_step_kernel<1><<<64, 256, 0, stream>>>(
                x, xbf, xproj, Wc, bperm, hbuf[t & 1], hbuf[(t + 1) & 1], c, t);
    } else {
        for (int t = 0; t < TT; ++t)
            lstm_step_kernel<0><<<64, 256, 0, stream>>>(
                x, xbf, xproj, Wc, bperm, hbuf[t & 1], hbuf[(t + 1) & 1], c, t);
    }
    // after 1024 steps (even count) the final h lives in hbuf[0]
    final_linear_kernel<<<(BB * 2 * LATN + 255) / 256, 256, 0, stream>>>(
        hbuf[0], w_lin, b_lin, out);
}
